// HNN1DWaveSeparable_51479478010707
// MI455X (gfx1250) — matrix operand order repaired and checked by reading, not compiled, not run on hardware
//
#include <hip/hip_runtime.h>
#include <cstdint>
#include <cstddef>

// ---------------------------------------------------------------------------
// HNN 1D wave separable: q_dot = -H[128+j, j], p_dot = -H[64+j, j]
// H = W1 [ diag(c1) + D1 W2 diag(c2) W2^T D1 ] W1^T   (per sample)
// Per sample: U = W2^T (d1 (.) Wx^T)  [512x64],  R = W2 (c2 (.) U) [512x64]
// Heavy part: two 512x512x64 GEMMs per sample -> v_wmma_f32_16x16x32_bf16.
// W1x bf16 tile is brought into LDS with global_load_async_to_lds_b128,
// overlapped with the forward pass; forward matvecs read bf16 W2 copies.
// ---------------------------------------------------------------------------

#define HID 512
#define NSP 64
#define DIN 192

// LDS layout (dynamic shared, no static __shared__ => segment base 0):
//   floats: zsm 256 | h1 512 | d1 512 | c1 512 | c2 512 | s2 512 | sumQ 64 | sumP 64
//   then bf16 tile[64][512]
#define TILE_LDS_OFF ((256 + 5 * 512 + 128) * 4)   // 11776 bytes
#define LDS_BYTES    (TILE_LDS_OFF + NSP * HID * 2)

typedef __attribute__((ext_vector_type(16))) __bf16         v16bf;
typedef __attribute__((ext_vector_type(8)))  __bf16         bf16x8;
typedef __attribute__((ext_vector_type(8)))  float          v8f;
typedef __attribute__((ext_vector_type(8)))  unsigned short u16x8;

__device__ __forceinline__ unsigned short f32_to_bf16_rne(float f) {
  union { float f; uint32_t u; } v; v.f = f;
  uint32_t u = v.u;
  uint32_t r = u + 0x7FFFu + ((u >> 16) & 1u);           // round-to-nearest-even
  if ((u & 0x7F800000u) == 0x7F800000u) r = u;           // NaN/Inf: truncate
  return (unsigned short)(r >> 16);
}

__device__ __forceinline__ float bf16_to_f32(unsigned short h) {
  union { uint32_t u; float f; } v; v.u = ((uint32_t)h) << 16; return v.f;
}

// A-fragment (16x32 bf16): lane row = lane&15; elements 0..15 -> K=k0..k0+15
// (k0 already includes the lane>=16 +16 shift).
__device__ __forceinline__ v16bf load_frag_split(const unsigned short* base) {
  bf16x8 lo = *reinterpret_cast<const bf16x8*>(base);
  bf16x8 hi = *reinterpret_cast<const bf16x8*>(base + 8);
  return __builtin_shufflevector(lo, hi, 0,1,2,3,4,5,6,7,8,9,10,11,12,13,14,15);
}

// B-fragment (32x16 bf16): lane col = lane&15; 16 consecutive K values.
__device__ __forceinline__ v16bf load_frag_cont(const unsigned short* base) {
  bf16x8 lo = *reinterpret_cast<const bf16x8*>(base);
  bf16x8 hi = *reinterpret_cast<const bf16x8*>(base + 8);
  return __builtin_shufflevector(lo, hi, 0,1,2,3,4,5,6,7,8,9,10,11,12,13,14,15);
}

// ---------------------------------------------------------------------------
// Kernel 0: one-time weight preprocessing into workspace.
// ---------------------------------------------------------------------------
__global__ void prep_weights(const float* __restrict__ W1,
                             const float* __restrict__ W2,
                             unsigned short* __restrict__ W2T_bf,  // [m][k]=W2[k][m]
                             unsigned short* __restrict__ W2_bf,   // [k][m]
                             unsigned short* __restrict__ W1x_bf,  // [j][k], j<64
                             float* __restrict__ PXQ,              // W1[j]*W1[64+j]
                             float* __restrict__ PXP) {            // W1[j]*W1[128+j]
  int gid = blockIdx.x * blockDim.x + threadIdx.x;
  if (gid < HID * HID) {
    int k = gid / HID, m = gid % HID;
    float w = W2[k * HID + m];
    W2_bf[k * HID + m]  = f32_to_bf16_rne(w);
    W2T_bf[m * HID + k] = f32_to_bf16_rne(w);
  }
  if (gid < NSP * HID) {
    int j = gid / HID, k = gid % HID;
    float wx = W1[j * HID + k];
    W1x_bf[gid] = f32_to_bf16_rne(wx);
    PXQ[gid] = wx * W1[(NSP     + j) * HID + k];
    PXP[gid] = wx * W1[(2 * NSP + j) * HID + k];
  }
}

// ---------------------------------------------------------------------------
// Main kernel: one workgroup (256 threads = 8 waves) per sample.
// ---------------------------------------------------------------------------
__global__ void __launch_bounds__(256)
hnn_kernel(const float* __restrict__ x, const float* __restrict__ q,
           const float* __restrict__ p, const float* __restrict__ W1,
           const float* __restrict__ b1, const float* __restrict__ b2,
           const float* __restrict__ W3,
           const unsigned short* __restrict__ W2T_bf,
           const unsigned short* __restrict__ W2_bf,
           const unsigned short* __restrict__ W1x_bf,
           const float* __restrict__ PXQ, const float* __restrict__ PXP,
           float* __restrict__ out, int B) {
  extern __shared__ char smem[];
  float* zsm  = reinterpret_cast<float*>(smem);     // 256 (192 used)
  float* h1   = zsm  + 256;                         // 512
  float* d1   = h1   + 512;                         // 512
  float* c1   = d1   + 512;                         // 512
  float* c2   = c1   + 512;                         // 512
  float* s2   = c2   + 512;                         // 512
  float* sumQ = s2   + 512;                         // 64
  float* sumP = sumQ + 64;                          // 64
  unsigned short* tile = reinterpret_cast<unsigned short*>(sumP + 64); // 64x512 bf16

  const int s = blockIdx.x;
  const int t = threadIdx.x;

  // ---- kick off async copy of W1x_bf (64 KB) into the LDS tile.
  // Per lane: LDS[vdst] = MEM[vaddr], 16 B, tracked by ASYNCcnt.
  {
    const unsigned short* gsrc = W1x_bf + (size_t)t * 8;    // 16 B per lane
    unsigned lds_addr = (unsigned)TILE_LDS_OFF + (unsigned)t * 16u;
    #pragma unroll
    for (int it = 0; it < 16; ++it) {
      asm volatile("global_load_async_to_lds_b128 %0, %1, off"
                   :: "v"(lds_addr), "v"(gsrc) : "memory");
      lds_addr += 256u * 16u;   // 4 KB per block-wide issue
      gsrc     += 256  * 8;
    }
  }

  if (t < 64) { sumQ[t] = 0.f; sumP[t] = 0.f; }
  if (t < DIN) {
    float v = (t < 64) ? x[s * NSP + t]
            : (t < 128) ? q[s * NSP + (t - 64)]
                        : p[s * NSP + (t - 128)];
    zsm[t] = v;
  }
  __syncthreads();

  // ---- forward: h1 = tanh(z W1 + b1)  (fp32 W1: only 192x512) ----
  for (int m = t; m < HID; m += 256) {
    float a = b1[m];
    #pragma unroll 4
    for (int i = 0; i < DIN; ++i) a += zsm[i] * W1[i * HID + m];
    float h = tanhf(a);
    h1[m] = h;
    d1[m] = 1.f - h * h;
  }
  __syncthreads();

  // ---- h2 = tanh(h1 W2 + b2); s2, c2  (bf16 W2: halves L2 traffic) ----
  for (int m = t; m < HID; m += 256) {
    float a = b2[m];
    #pragma unroll 4
    for (int k = 0; k < HID; ++k) a += h1[k] * bf16_to_f32(W2_bf[k * HID + m]);
    float h  = tanhf(a);
    float dh = 1.f - h * h;
    float w3 = W3[m];
    s2[m] = dh * w3;
    c2[m] = -2.f * h * dh * w3;
  }
  __syncthreads();

  // ---- v1 = W2 s2; c1 ----
  for (int k = t; k < HID; k += 256) {
    float a = 0.f;
    const unsigned short* w2row = W2_bf + k * HID;
    #pragma unroll 4
    for (int m = 0; m < HID; ++m) a += bf16_to_f32(w2row[m]) * s2[m];
    c1[k] = -2.f * h1[k] * d1[k] * a;
  }
  __syncthreads();

  // ---- async tile landed long ago; fence it, then scale in place by d1 ----
  asm volatile("s_wait_asynccnt 0x0" ::: "memory");
  __syncthreads();
  for (int e = t * 8; e < NSP * HID; e += 256 * 8) {
    int k0 = e & 511;
    u16x8 v = *reinterpret_cast<u16x8*>(tile + e);
    u16x8 o;
    #pragma unroll
    for (int r = 0; r < 8; ++r)
      o[r] = f32_to_bf16_rne(bf16_to_f32(v[r]) * d1[k0 + r]);
    *reinterpret_cast<u16x8*>(tile + e) = o;
  }
  // ---- first (diag(c1)) term, independent of the GEMMs ----
  {
    int j = t & 63, seg = t >> 6;
    const float* pq  = PXQ + j * HID + seg * 128;
    const float* pp  = PXP + j * HID + seg * 128;
    const float* c1s = c1 + seg * 128;
    float sq = 0.f, sp = 0.f;
    #pragma unroll 4
    for (int k = 0; k < 128; ++k) { sq += c1s[k] * pp[k]; sp += c1s[k] * pq[k]; }
    atomicAdd(&sumQ[j], sq);
    atomicAdd(&sumP[j], sp);
  }
  __syncthreads();

  const int wv   = t >> 5;
  const int lane = t & 31;
  const int lr   = lane & 15;
  const int hi   = (lane >> 4) & 1;

  v8f acc[4][4];
  #pragma unroll
  for (int a = 0; a < 4; ++a)
    #pragma unroll
    for (int b = 0; b < 4; ++b)
      acc[a][b] = (v8f){0.f, 0.f, 0.f, 0.f, 0.f, 0.f, 0.f, 0.f};

  // ================= Stage 1: U = W2^T * Ahat  (wave owns rows wv*64..+63) ==
  {
    const unsigned short* Arow[4];
    #pragma unroll
    for (int mi = 0; mi < 4; ++mi)
      Arow[mi] = W2T_bf + (size_t)(wv * 64 + mi * 16 + lr) * HID;
    for (int kb = 0; kb < 16; ++kb) {
      const int ka  = kb * 32 + hi * 16;  // A-frag K base (16 contiguous)
      const int kbb = kb * 32 + hi * 16;  // B-frag K base (16 contiguous)
      v16bf af[4], bfr[4];
      #pragma unroll
      for (int mi = 0; mi < 4; ++mi) af[mi]  = load_frag_split(Arow[mi] + ka);
      #pragma unroll
      for (int ni = 0; ni < 4; ++ni) bfr[ni] = load_frag_cont(tile + (ni * 16 + lr) * HID + kbb);
      #pragma unroll
      for (int mi = 0; mi < 4; ++mi)
        #pragma unroll
        for (int ni = 0; ni < 4; ++ni)
          acc[mi][ni] = __builtin_amdgcn_wmma_f32_16x16x32_bf16(
              false, af[mi], false, bfr[ni], (short)0, acc[mi][ni], false, false);
    }
  }
  __syncthreads();  // all waves done reading Ahat

  // ---- tile <- bf16(c2 (.) U), layout Uhat[j][m] (K=m contiguous) ----
  #pragma unroll
  for (int mi = 0; mi < 4; ++mi) {
    int m0 = wv * 64 + mi * 16 + hi * 8;
    #pragma unroll
    for (int ni = 0; ni < 4; ++ni) {
      int j = ni * 16 + lr;
      u16x8 pk;
      #pragma unroll
      for (int r = 0; r < 8; ++r)
        pk[r] = f32_to_bf16_rne(acc[mi][ni][r] * c2[m0 + r]);
      *reinterpret_cast<u16x8*>(tile + j * HID + m0) = pk;
      acc[mi][ni] = (v8f){0.f, 0.f, 0.f, 0.f, 0.f, 0.f, 0.f, 0.f};
    }
  }
  __syncthreads();

  // ================= Stage 2: R = W2 * Uhat =================================
  {
    const unsigned short* Arow[4];
    #pragma unroll
    for (int mi = 0; mi < 4; ++mi)
      Arow[mi] = W2_bf + (size_t)(wv * 64 + mi * 16 + lr) * HID;
    for (int kb = 0; kb < 16; ++kb) {
      const int ka  = kb * 32 + hi * 16;
      const int kbb = kb * 32 + hi * 16;
      v16bf af[4], bfr[4];
      #pragma unroll
      for (int mi = 0; mi < 4; ++mi) af[mi]  = load_frag_split(Arow[mi] + ka);
      #pragma unroll
      for (int ni = 0; ni < 4; ++ni) bfr[ni] = load_frag_cont(tile + (ni * 16 + lr) * HID + kbb);
      #pragma unroll
      for (int mi = 0; mi < 4; ++mi)
        #pragma unroll
        for (int ni = 0; ni < 4; ++ni)
          acc[mi][ni] = __builtin_amdgcn_wmma_f32_16x16x32_bf16(
              false, af[mi], false, bfr[ni], (short)0, acc[mi][ni], false, false);
    }
  }

  // ---- second-term epilogue from R accumulators ----
  #pragma unroll
  for (int ni = 0; ni < 4; ++ni) {
    int j = ni * 16 + lr;
    float sq = 0.f, sp = 0.f;
    #pragma unroll
    for (int mi = 0; mi < 4; ++mi) {
      int i0 = wv * 64 + mi * 16 + hi * 8;
      const float* w1p = W1 + (size_t)(2 * NSP + j) * HID + i0;  // row 128+j
      const float* w1q = W1 + (size_t)(NSP     + j) * HID + i0;  // row  64+j
      #pragma unroll
      for (int r = 0; r < 8; ++r) {
        float dval = d1[i0 + r] * acc[mi][ni][r];
        sq += dval * w1p[r];
        sp += dval * w1q[r];
      }
    }
    atomicAdd(&sumQ[j], sq);
    atomicAdd(&sumP[j], sp);
  }
  __syncthreads();

  if (t < 64) {
    out[(size_t)s * NSP + t]                   = -sumQ[t];  // q_dot = -dH_dp_dx
    out[(size_t)B * NSP + (size_t)s * NSP + t] = -sumP[t];  // p_dot = -dH_dq_dx
  }
}

// ---------------------------------------------------------------------------
extern "C" void kernel_launch(void* const* d_in, const int* in_sizes, int n_in,
                              void* d_out, int out_size, void* d_ws, size_t ws_size,
                              hipStream_t stream) {
  const float* x  = (const float*)d_in[0];
  const float* q  = (const float*)d_in[1];
  const float* p  = (const float*)d_in[2];
  const float* W1 = (const float*)d_in[3];
  const float* b1 = (const float*)d_in[4];
  const float* W2 = (const float*)d_in[5];
  const float* b2 = (const float*)d_in[6];
  const float* W3 = (const float*)d_in[7];
  // d_in[8] = b3: unused (constants vanish under differentiation)

  const int B = in_sizes[0] / NSP;

  unsigned short* W2T_bf = (unsigned short*)d_ws;           // 512*512 bf16
  unsigned short* W2_bf  = W2T_bf + HID * HID;              // 512*512 bf16
  unsigned short* W1x_bf = W2_bf  + HID * HID;              // 64*512 bf16
  float* PXQ = (float*)(W1x_bf + NSP * HID);                // 64*512 f32
  float* PXP = PXQ + NSP * HID;                             // 64*512 f32

  prep_weights<<<(HID * HID) / 256, 256, 0, stream>>>(W1, W2, W2T_bf, W2_bf,
                                                      W1x_bf, PXQ, PXP);

  hnn_kernel<<<B, 256, (size_t)LDS_BYTES, stream>>>(x, q, p, W1, b1, b2, W3,
                                                    W2T_bf, W2_bf, W1x_bf,
                                                    PXQ, PXP, (float*)d_out, B);
}
